// UCCAEncoder_50938312130561
// MI455X (gfx1250) — matrix operand order repaired and checked by reading, not compiled, not run on hardware
//
#include <hip/hip_runtime.h>
#include <hip/hip_bf16.h>

// ---------------------------------------------------------------------------
// UCCA encoder: 3x EdgeConv (per-edge MLP + segment_max) + FFN on selected rows
// f16 WMMA (v_wmma_f32_16x16x32_f16), fp32 accumulate.
//  - edges counting-sorted by dst once per call -> tile-local segmented max,
//    ~13x fewer global fp32 atomics
//  - M=64 edges per block (4x16 subtiles), B fragments reused across subtiles
//  - 128 KB static LDS, triple-aliased: GEMM1 A-frags -> GEMM2 Y-frags -> fp32 D tile
// ---------------------------------------------------------------------------

typedef __attribute__((ext_vector_type(16))) _Float16 v16h;
typedef __attribute__((ext_vector_type(4)))  _Float16 v4h;
typedef __attribute__((ext_vector_type(8)))  float    v8f;

#define HDIM 512
#define NT   32   // HDIM / 16 column tiles
#define SUB  4    // 16-edge subtiles per block (M = 16*SUB)
#define NNODE 16384

__device__ __forceinline__ v8f wmma_f16(v16h a, v16h b, v8f c) {
  return __builtin_amdgcn_wmma_f32_16x16x32_f16(false, a, false, b, (short)0, c, false, false);
}

// A-matrix (16x32 f16) fragment offset for element (row m in 0..15, k):
//   addr = ((kt*32 + lane) * 16 + e), lane = m + 16*hi,
//   K(e,lane) = (e<8 ? e : e+8) + (lane>=16 ? 8 : 0)     [ISA 05_wmma 7.12.2]
__device__ __forceinline__ int a_frag_off(int m, int k) {
  const int kt = k >> 5;
  const int kk = k & 31;
  const int hi = ((kk & 15) >= 8) ? 1 : 0;
  const int base = kk - (hi << 3);
  const int e = (base < 8) ? base : (base - 8);
  const int L = m + (hi << 4);
  return (((kt << 5) + L) << 4) + e;
}

// fp32 row-major W[K, Nn] -> f16 B-fragment layout:
//   out[(((kt*(Nn/16)+nt)*32 + lane)*16 + e)] = W[kt*32 + (e<8 ? e : e+8) + 8*(lane>=16)][nt*16 + lane%16]
__global__ void convert_weight(const float* __restrict__ W, _Float16* __restrict__ out,
                               int K, int Nn) {
  const int idx = blockIdx.x * blockDim.x + threadIdx.x;
  if (idx >= K * Nn) return;
  const int e    = idx & 15;
  const int lane = (idx >> 4) & 31;
  const int rest = idx >> 9;
  const int ntiles = Nn >> 4;
  const int nt = rest % ntiles;
  const int kt = rest / ntiles;
  const int k = (kt << 5) + ((e < 8) ? e : (e + 8)) + ((lane >= 16) ? 8 : 0);
  const int n = (nt << 4) + (lane & 15);
  out[idx] = (_Float16)W[(size_t)k * Nn + n];
}

__global__ void init_neg_inf(float* __restrict__ p, int n) {
  for (int i = blockIdx.x * blockDim.x + threadIdx.x; i < n; i += gridDim.x * blockDim.x)
    p[i] = -__builtin_inff();
}

__global__ void zero_int(int* __restrict__ p, int n) {
  for (int i = blockIdx.x * blockDim.x + threadIdx.x; i < n; i += gridDim.x * blockDim.x)
    p[i] = 0;
}

// segment_max fill: empty segments were -inf -> 0; then relu => max(agg, 0)
__global__ void relu_from_agg(const float* __restrict__ agg, float* __restrict__ h, int n) {
  for (int i = blockIdx.x * blockDim.x + threadIdx.x; i < n; i += gridDim.x * blockDim.x)
    h[i] = fmaxf(agg[i], 0.0f);
}

// ---- counting sort of edges by destination (edge list fixed across layers) ----
__global__ void hist_dst(const long long* __restrict__ dst, int* __restrict__ deg, int E) {
  const int e = blockIdx.x * blockDim.x + threadIdx.x;
  if (e < E) atomicAdd(&deg[(int)dst[e]], 1);
}

// exclusive scan of 16384 bins with one 1024-thread block -> cursor
__global__ __launch_bounds__(1024)
void scan_cursor(const int* __restrict__ deg, int* __restrict__ cursor) {
  __shared__ int part[1024];
  const int t = threadIdx.x;
  const int base = t * 16;
  int local[16];
  int s = 0;
  #pragma unroll
  for (int i = 0; i < 16; ++i) { local[i] = s; s += deg[base + i]; }
  part[t] = s;
  for (int off = 1; off < 1024; off <<= 1) {
    __syncthreads();
    const int add = (t >= off) ? part[t - off] : 0;
    const int v = part[t];
    __syncthreads();
    part[t] = v + add;               // inclusive scan of chunk sums
  }
  __syncthreads();
  const int prefix = (t == 0) ? 0 : part[t - 1];
  #pragma unroll
  for (int i = 0; i < 16; ++i) cursor[base + i] = prefix + local[i];
}

__global__ void scatter_perm(const long long* __restrict__ dst, int* __restrict__ cursor,
                             int* __restrict__ perm, int E) {
  const int e = blockIdx.x * blockDim.x + threadIdx.x;
  if (e < E) {
    const int pos = atomicAdd(&cursor[(int)dst[e]], 1);
    perm[pos] = e;                   // intra-bin order arbitrary; max is order-invariant
  }
}

// ---------------------------------------------------------------------------
// Block = 512 threads = 16 waves; handles 16*SUB dst-sorted edges.
// sBuf aliases: [SUB][32kt][512] f16 A-frags (128KB) -> [SUB][16kt][512] f16
// Y-frags (64KB) -> [64][512] fp32 D tile (128KB).
// ---------------------------------------------------------------------------
__global__ __launch_bounds__(512)
void edge_conv_kernel(const float* __restrict__ h,
                      const long long* __restrict__ eidx,   // [2, E]: src then dst
                      const int* __restrict__ perm,         // edges sorted by dst
                      long long E,
                      const _Float16* __restrict__ w1c, const float* __restrict__ b1,
                      const _Float16* __restrict__ w2c, const float* __restrict__ b2,
                      float* __restrict__ agg) {
  __shared__ _Float16 sBuf[SUB * 32 * 512];   // 128 KB (CDNA5: 320 KB/WGP)
  __shared__ int sDst[SUB * 16];

  const int lane = threadIdx.x & 31;
  const int wave = threadIdx.x >> 5;
  const long long ebase = (long long)blockIdx.x * (16 * SUB);

  // ---- gather (float4 loads, packed 4xf16 LDS stores)
  #pragma unroll
  for (int s = 0; s < SUB; ++s) {
    const int m = wave;
    const int eid = perm[ebase + s * 16 + m];
    const int srow = (int)eidx[eid];        // x_j = source
    const int drow = (int)eidx[E + eid];    // x_i = target
    if (lane == 0) sDst[s * 16 + m] = drow;
    const float* __restrict__ pi = h + (size_t)drow * HDIM;
    const float* __restrict__ pj = h + (size_t)srow * HDIM;
    _Float16* sA = sBuf + s * (32 * 512);
    for (int f0 = lane * 4; f0 < HDIM; f0 += 128) {
      const float4 xi = *(const float4*)&pi[f0];
      const float4 xj = *(const float4*)&pj[f0];
      v4h hi4, hd4;
      hi4[0] = (_Float16)xi.x; hi4[1] = (_Float16)xi.y;
      hi4[2] = (_Float16)xi.z; hi4[3] = (_Float16)xi.w;
      hd4[0] = (_Float16)(xj.x - xi.x); hd4[1] = (_Float16)(xj.y - xi.y);
      hd4[2] = (_Float16)(xj.z - xi.z); hd4[3] = (_Float16)(xj.w - xi.w);
      // k..k+3 (4-aligned) share kt/hi and have consecutive e -> one b64 store
      *(v4h*)&sA[a_frag_off(m, f0)]        = hi4;   // ef[0:512]    = x_i
      *(v4h*)&sA[a_frag_off(m, HDIM + f0)] = hd4;   // ef[512:1024] = x_j - x_i
    }
  }
  __syncthreads();

  const int nt0   = wave * 2;                 // 16 waves x 2 tiles = 32 col tiles
  const int mbase = (lane >= 16) ? 8 : 0;     // C/D row base (16x16 f32 layout)
  const int ncol  = lane & 15;

  // ---- GEMM1: K = 1024 (32 chunks); B loaded once per chunk, reused SUB times
  v8f acc[SUB][2];
  #pragma unroll
  for (int s = 0; s < SUB; ++s) { acc[s][0] = (v8f){}; acc[s][1] = (v8f){}; }

  for (int kt = 0; kt < 32; ++kt) {
    const v16h bf0 = *(const v16h*)&w1c[(((size_t)(kt * NT + nt0)     << 5) + lane) << 4];
    const v16h bf1 = *(const v16h*)&w1c[(((size_t)(kt * NT + nt0 + 1) << 5) + lane) << 4];
    #pragma unroll
    for (int s = 0; s < SUB; ++s) {
      const v16h a = *(const v16h*)&sBuf[s * (32 * 512) + ((((kt << 5) + lane) << 4))];
      acc[s][0] = wmma_f16(a, bf0, acc[s][0]);
      acc[s][1] = wmma_f16(a, bf1, acc[s][1]);
    }
  }
  __syncthreads();   // all waves done reading GEMM1 A-frags; safe to alias

  // ---- epilogue 1: +b1, relu, store f16 Y-frags (K=512) into aliased sBuf
  #pragma unroll
  for (int t = 0; t < 2; ++t) {
    const int col = (nt0 + t) * 16 + ncol;
    const float bias = b1[col];
    const int kt2 = col >> 5, kk = col & 31;
    const int hi = ((kk & 15) >= 8) ? 1 : 0;
    const int basek = kk - (hi << 3);
    const int e2 = (basek < 8) ? basek : (basek - 8);
    #pragma unroll
    for (int s = 0; s < SUB; ++s) {
      _Float16* sY = sBuf + s * (16 * 512);
      #pragma unroll
      for (int r = 0; r < 8; ++r) {
        float v = acc[s][t][r] + bias;
        v = v > 0.0f ? v : 0.0f;
        const int L2 = (r + mbase) + (hi << 4);
        sY[(((kt2 << 5) + L2) << 4) + e2] = (_Float16)v;
      }
    }
  }
  __syncthreads();

  // ---- GEMM2: K = 512 (16 chunks); B reused across subtiles
  v8f d[SUB][2];
  #pragma unroll
  for (int s = 0; s < SUB; ++s) { d[s][0] = (v8f){}; d[s][1] = (v8f){}; }

  for (int kt = 0; kt < 16; ++kt) {
    const v16h bf0 = *(const v16h*)&w2c[(((size_t)(kt * NT + nt0)     << 5) + lane) << 4];
    const v16h bf1 = *(const v16h*)&w2c[(((size_t)(kt * NT + nt0 + 1) << 5) + lane) << 4];
    #pragma unroll
    for (int s = 0; s < SUB; ++s) {
      const v16h a = *(const v16h*)&sBuf[s * (16 * 512) + ((((kt << 5) + lane) << 4))];
      d[s][0] = wmma_f16(a, bf0, d[s][0]);
      d[s][1] = wmma_f16(a, bf1, d[s][1]);
    }
  }
  __syncthreads();   // GEMM2 done reading Y-frags; alias sBuf as fp32 D tile

  // ---- epilogue 2: +b2 into LDS fp32 tile [64][512]
  float* sD = (float*)sBuf;
  #pragma unroll
  for (int t = 0; t < 2; ++t) {
    const int col = (nt0 + t) * 16 + ncol;
    const float bias = b2[col];
    #pragma unroll
    for (int s = 0; s < SUB; ++s) {
      #pragma unroll
      for (int r = 0; r < 8; ++r)
        sD[(size_t)(s * 16 + r + mbase) * HDIM + col] = d[s][t][r] + bias;
    }
  }
  __syncthreads();

  // ---- tile-local segmented max over dst-sorted rows; atomic only on dst change
  {
    const int col = threadIdx.x;           // 512 threads = 512 columns
    int cur = sDst[0];
    float run = -__builtin_inff();
    for (int row = 0; row < 16 * SUB; ++row) {
      const int dr = sDst[row];
      const float v = sD[(size_t)row * HDIM + col];
      if (dr != cur) {
        atomicMax(&agg[(size_t)cur * HDIM + col], run);
        cur = dr;
        run = v;
      } else {
        run = fmaxf(run, v);
      }
    }
    atomicMax(&agg[(size_t)cur * HDIM + col], run);
  }
}

// FFN only on the B*SEL = 4096 selected rows; one block = one 16-row tile.
__global__ __launch_bounds__(512)
void ffn_select_kernel(const float* __restrict__ h,
                       const long long* __restrict__ sel,   // [B, 512]
                       const _Float16* __restrict__ w1c, const float* __restrict__ b1,
                       const _Float16* __restrict__ w2c, const float* __restrict__ b2,
                       float* __restrict__ out) {
  __shared__ _Float16 sX[16 * 512];
  __shared__ _Float16 sY[16 * 512];

  const int lane = threadIdx.x & 31;
  const int wave = threadIdx.x >> 5;
  const int gr0 = blockIdx.x * 16;

  { // gather selected rows (float4 vectorized)
    const int m = wave;
    const int grow = gr0 + m;            // 0..4095
    const int bb = grow >> 9;            // SEL = 512
    const int jj = grow & 511;
    const int row = bb * 2048 + (int)sel[(size_t)bb * 512 + jj];
    const float* __restrict__ p = h + (size_t)row * HDIM;
    for (int f0 = lane * 4; f0 < HDIM; f0 += 128) {
      const float4 xv = *(const float4*)&p[f0];
      v4h hv;
      hv[0] = (_Float16)xv.x; hv[1] = (_Float16)xv.y;
      hv[2] = (_Float16)xv.z; hv[3] = (_Float16)xv.w;
      *(v4h*)&sX[a_frag_off(m, f0)] = hv;
    }
  }
  __syncthreads();

  const int nt0   = wave * 2;
  const int mbase = (lane >= 16) ? 8 : 0;
  const int ncol  = lane & 15;

  v8f acc0 = {}; v8f acc1 = {};
  for (int kt = 0; kt < 16; ++kt) {
    const v16h a   = *(const v16h*)&sX[(((kt << 5) + lane) << 4)];
    const v16h bf0 = *(const v16h*)&w1c[(((size_t)(kt * NT + nt0)     << 5) + lane) << 4];
    const v16h bf1 = *(const v16h*)&w1c[(((size_t)(kt * NT + nt0 + 1) << 5) + lane) << 4];
    acc0 = wmma_f16(a, bf0, acc0);
    acc1 = wmma_f16(a, bf1, acc1);
  }
  #pragma unroll
  for (int t = 0; t < 2; ++t) {
    const v8f acc = t ? acc1 : acc0;
    const int col = (nt0 + t) * 16 + ncol;
    const float bias = b1[col];
    const int kt2 = col >> 5, kk = col & 31;
    const int hi = ((kk & 15) >= 8) ? 1 : 0;
    const int basek = kk - (hi << 3);
    const int e2 = (basek < 8) ? basek : (basek - 8);
    #pragma unroll
    for (int r = 0; r < 8; ++r) {
      float v = acc[r] + bias;
      v = v > 0.0f ? v : 0.0f;
      const int L2 = (r + mbase) + (hi << 4);
      sY[(((kt2 << 5) + L2) << 4) + e2] = (_Float16)v;
    }
  }
  __syncthreads();

  v8f d0 = {}; v8f d1 = {};
  for (int kt = 0; kt < 16; ++kt) {
    const v16h a   = *(const v16h*)&sY[(((kt << 5) + lane) << 4)];
    const v16h bf0 = *(const v16h*)&w2c[(((size_t)(kt * NT + nt0)     << 5) + lane) << 4];
    const v16h bf1 = *(const v16h*)&w2c[(((size_t)(kt * NT + nt0 + 1) << 5) + lane) << 4];
    d0 = wmma_f16(a, bf0, d0);
    d1 = wmma_f16(a, bf1, d1);
  }
  #pragma unroll
  for (int t = 0; t < 2; ++t) {
    const v8f acc = t ? d1 : d0;
    const int col = (nt0 + t) * 16 + ncol;
    const float bias = b2[col];
    #pragma unroll
    for (int r = 0; r < 8; ++r)
      out[(size_t)(gr0 + r + mbase) * HDIM + col] = acc[r] + bias;
  }
}

extern "C" void kernel_launch(void* const* d_in, const int* in_sizes, int n_in,
                              void* d_out, int out_size, void* d_ws, size_t ws_size,
                              hipStream_t stream) {
  (void)in_sizes; (void)n_in; (void)out_size; (void)ws_size;
  const long long Bq = 8, S = 2048, E = 262144, SELn = 512;
  const long long N = Bq * S;

  const float*     x    = (const float*)d_in[0];
  const long long* eidx = (const long long*)d_in[1];     // int64 [2,E]
  const long long* sel  = (const long long*)d_in[2];     // int64 [B,SEL]
  // d_in[3] = edge_label: unused by reference
  const float* cw1[3] = {(const float*)d_in[4],  (const float*)d_in[8],  (const float*)d_in[12]};
  const float* cb1[3] = {(const float*)d_in[5],  (const float*)d_in[9],  (const float*)d_in[13]};
  const float* cw2[3] = {(const float*)d_in[6],  (const float*)d_in[10], (const float*)d_in[14]};
  const float* cb2[3] = {(const float*)d_in[7],  (const float*)d_in[11], (const float*)d_in[15]};
  const float* fw1 = (const float*)d_in[16];
  const float* fb1 = (const float*)d_in[17];
  const float* fw2 = (const float*)d_in[18];
  const float* fb2 = (const float*)d_in[19];

  // workspace layout
  char* ws = (char*)d_ws;
  size_t off = 0;
  float* hbuf = (float*)(ws + off); off += (size_t)N * HDIM * 4;   // 32 MB
  float* agg  = (float*)(ws + off); off += (size_t)N * HDIM * 4;   // 32 MB
  _Float16* w1c[3]; _Float16* w2c[3];
  for (int l = 0; l < 3; ++l) { w1c[l] = (_Float16*)(ws + off); off += 1024ull * 512 * 2; }
  for (int l = 0; l < 3; ++l) { w2c[l] = (_Float16*)(ws + off); off += 512ull * 512 * 2; }
  _Float16* fw1c = (_Float16*)(ws + off); off += 512ull * 512 * 2;
  _Float16* fw2c = (_Float16*)(ws + off); off += 512ull * 512 * 2;
  int* deg    = (int*)(ws + off); off += (size_t)NNODE * 4;
  int* cursor = (int*)(ws + off); off += (size_t)NNODE * 4;
  int* perm   = (int*)(ws + off); off += (size_t)E * 4;

  // per-call weight down-convert into B-fragment layout
  for (int l = 0; l < 3; ++l) {
    convert_weight<<<(1024 * 512 + 255) / 256, 256, 0, stream>>>(cw1[l], w1c[l], 1024, 512);
    convert_weight<<<(512 * 512 + 255) / 256, 256, 0, stream>>>(cw2[l], w2c[l], 512, 512);
  }
  convert_weight<<<(512 * 512 + 255) / 256, 256, 0, stream>>>(fw1, fw1c, 512, 512);
  convert_weight<<<(512 * 512 + 255) / 256, 256, 0, stream>>>(fw2, fw2c, 512, 512);

  // counting sort of edges by dst (shared by all 3 conv layers)
  zero_int<<<64, 256, 0, stream>>>(deg, NNODE);
  hist_dst<<<(int)(E / 256), 256, 0, stream>>>(eidx + E, deg, (int)E);
  scan_cursor<<<1, 1024, 0, stream>>>(deg, cursor);
  scatter_perm<<<(int)(E / 256), 256, 0, stream>>>(eidx + E, cursor, perm, (int)E);

  const int nelem = (int)(N * HDIM);
  const float* cur = x;
  for (int l = 0; l < 3; ++l) {
    init_neg_inf<<<2048, 256, 0, stream>>>(agg, nelem);
    edge_conv_kernel<<<(int)(E / (16 * SUB)), 512, 0, stream>>>(cur, eidx, perm, E,
                                                                w1c[l], cb1[l], w2c[l], cb2[l], agg);
    relu_from_agg<<<2048, 256, 0, stream>>>(agg, hbuf, nelem);
    cur = hbuf;
  }
  ffn_select_kernel<<<(int)(Bq * SELn / 16), 512, 0, stream>>>(hbuf, sel,
                                                               fw1c, fb1, fw2c, fb2,
                                                               (float*)d_out);
}
